// MHSA_22531398435392
// MI455X (gfx1250) — compile-verified, matrix operand order checked by reading, not run on hardware
//
#include <hip/hip_runtime.h>

// ---------------------------------------------------------------------------
// MHSA forward for MI455X (gfx1250), bf16 WMMA + fp32 accumulate, flash-attn,
// with Tensor Data Mover (TDM) double-buffered K/V staging into LDS.
// B=4, P=4096, C=256, H=4, D=64, SCALE = D^-0.5 = 0.125
// ---------------------------------------------------------------------------

#define BATCH 4
#define SEQ   4096
#define CH    256
#define NH    4
#define HD    64
#define SCALE 0.125f

typedef __attribute__((ext_vector_type(16))) __bf16 v16bf;
typedef __attribute__((ext_vector_type(8)))  __bf16 v8bf;
typedef __attribute__((ext_vector_type(8)))  float  v8f;
typedef __attribute__((ext_vector_type(4)))  float  v4f;
typedef __attribute__((ext_vector_type(4)))  unsigned int v4u;
typedef __attribute__((ext_vector_type(8)))  int    v8i;
typedef __attribute__((ext_vector_type(4)))  int    v4i;

#define WMMA_BF16(A, B, C) \
  __builtin_amdgcn_wmma_f32_16x16x32_bf16(false, (A), false, (B), (short)0, (C), false, false)

// CDNA5 16-bit A/B fragment K-layout: lanes<16 hold K = koff+{0..7} and
// 16+koff+{0..7}; lanes>=16 use koff=8. Both 8-element groups are contiguous
// in memory when the K dimension is contiguous -> two b128 loads.
static __device__ __forceinline__ v16bf load_frag16(const __bf16* base, int koff) {
  v8bf lo = *(const v8bf*)(base + koff);
  v8bf hi = *(const v8bf*)(base + 16 + koff);
  v16bf r;
#pragma unroll
  for (int i = 0; i < 8; ++i) { r[i] = lo[i]; r[8 + i] = hi[i]; }
  return r;
}

// Same fragment from fp32 source (convert to bf16 on the fly).
static __device__ __forceinline__ v16bf load_frag16_f32(const float* base, int koff) {
  v4f f0 = *(const v4f*)(base + koff);
  v4f f1 = *(const v4f*)(base + koff + 4);
  v4f f2 = *(const v4f*)(base + 16 + koff);
  v4f f3 = *(const v4f*)(base + 16 + koff + 4);
  v16bf r;
#pragma unroll
  for (int i = 0; i < 4; ++i) {
    r[i]      = (__bf16)f0[i];
    r[4 + i]  = (__bf16)f1[i];
    r[8 + i]  = (__bf16)f2[i];
    r[12 + i] = (__bf16)f3[i];
  }
  return r;
}

// ---------------------------------------------------------------------------
// Tensor Data Mover: async 2-D tile load (bf16 elements) global -> LDS.
// Builds the D# descriptor per CDNA5 ISA:
//  group0: [1:0]=count=1, [63:32]=lds_addr, [120:64]=global_addr, [127:126]=2
//  group1: data_size=1 (2B) @ [17:16]; tensor_dim0 @ [79:48];
//          tensor_dim1 @ [111:80]; tile_dim0 @ [127:112]; tile_dim1 @ [143:128];
//          tensor_dim0_stride @ [207:160].  groups 2/3 zero (2-D tensor).
// Tracked on TENSORcnt; wait with s_wait_tensorcnt.
// ---------------------------------------------------------------------------
static __device__ __forceinline__ void tdm_load_2d_bf16(
    unsigned int lds_off, const void* gaddr,
    unsigned int dim0, unsigned int dim1,
    unsigned int tile0, unsigned int tile1, unsigned int stride0) {
  unsigned long long ga = (unsigned long long)gaddr;
  v4u g0 = {1u,                                   // count = 1 valid descriptor
            lds_off,                              // LDS byte address
            (unsigned int)ga,                     // global_addr[31:0]
            (((unsigned int)(ga >> 32)) & 0x01FFFFFFu) | 0x80000000u}; // type=2
  v8i g1 = {(int)0x00010000u,                           // data_size = 2 bytes
            (int)((dim0 & 0xFFFFu) << 16),              // tensor_dim0[15:0]
            (int)((dim0 >> 16) | ((dim1 & 0xFFFFu) << 16)),
            (int)((dim1 >> 16) | (tile0 << 16)),        // tile_dim0
            (int)tile1,                                 // tile_dim1 (tile_dim2=0)
            (int)stride0,                               // tensor_dim0_stride[31:0]
            0, 0};
  v4i z4 = {0, 0, 0, 0};
#if defined(__clang_major__) && (__clang_major__ >= 23)
  v8i z8 = {0, 0, 0, 0, 0, 0, 0, 0};
  __builtin_amdgcn_tensor_load_to_lds(g0, g1, z4, z4, z8, 0);
#else
  __builtin_amdgcn_tensor_load_to_lds(g0, g1, z4, z4, 0);
#endif
}

// ---------------------------------------------------------------------------
// Phase 1: Q/K/V projections.  grid=(256, 4, 3) block=128.
//  z selects projection; block tile = 64 rows x 64 cols; wave tile = 16x64.
//  Q (pre-scaled) and K stored row-major [b,h,p,d] bf16; V stored transposed
//  [b,h,d,p] bf16 so the attention V B-fragments are contiguous b128 loads.
// ---------------------------------------------------------------------------
__global__ __launch_bounds__(128) void qkv_kernel(
    const float* __restrict__ x,
    const float* __restrict__ Wq, const float* __restrict__ bq,
    const float* __restrict__ Wk, const float* __restrict__ bk,
    const float* __restrict__ Wv, const float* __restrict__ bv,
    __bf16* __restrict__ Qb, __bf16* __restrict__ Kb, __bf16* __restrict__ Vt) {
  const int proj = blockIdx.z;
  const float* W    = (proj == 0) ? Wq : (proj == 1) ? Wk : Wv;
  const float* bias = (proj == 0) ? bq : (proj == 1) ? bk : bv;

  const int tid  = threadIdx.x;
  const int lane = tid & 31;
  const int wave = tid >> 5;
  const int l16  = lane & 15;
  const int koff = (lane < 16) ? 0 : 8;
  const int m0   = blockIdx.x * 64 + wave * 16;
  const int n0   = blockIdx.y * 64;

  __shared__ __attribute__((aligned(16))) __bf16 wt[64][32];  // W tile [n][k]

  v8f acc[4];
#pragma unroll
  for (int t = 0; t < 4; ++t)
#pragma unroll
    for (int r = 0; r < 8; ++r) acc[t][r] = 0.0f;

  const int    M    = m0 + l16;              // A-frag row for this lane
  const float* arow = x + (size_t)M * CH;

  for (int k0 = 0; k0 < CH; k0 += 32) {
    __syncthreads();
#pragma unroll
    for (int i = 0; i < 16; ++i) {           // cooperative stage: 32x64 fp32
      int idx = tid + i * 128;
      int n = idx & 63, k = idx >> 6;
      wt[n][k] = (__bf16)W[(size_t)(k0 + k) * CH + n0 + n];
    }
    __syncthreads();

    v16bf a = load_frag16_f32(arow + k0, koff);
#pragma unroll
    for (int t = 0; t < 4; ++t) {
      v16bf bfrag = load_frag16(&wt[t * 16 + l16][0], koff);
      acc[t] = WMMA_BF16(a, bfrag, acc[t]);
    }
  }

  // Epilogue. C-tile: VGPR r -> row m0 + r + 8*(lane>=16); lane%16 -> col.
  const int rowbase = m0 + ((lane >> 4) << 3);
  const int bidx    = rowbase >> 12;         // row / 4096 (tiles never straddle)
  const int p0      = rowbase & (SEQ - 1);

#pragma unroll
  for (int t = 0; t < 4; ++t) {
    const int   col = n0 + t * 16 + l16;
    const float bb  = bias[col];
    const int   h   = col >> 6;
    const int   d   = col & 63;
    if (proj == 2) {  // V transposed: 8 consecutive p -> one b128 store
      v8bf pack;
#pragma unroll
      for (int r = 0; r < 8; ++r) pack[r] = (__bf16)(acc[t][r] + bb);
      __bf16* dst = Vt + ((size_t)((bidx * NH + h) * HD + d)) * SEQ + p0;
      *(v8bf*)dst = pack;
    } else {
      __bf16* base = ((proj == 0) ? Qb : Kb) +
                     ((size_t)(bidx * NH + h)) * SEQ * HD + d;
#pragma unroll
      for (int r = 0; r < 8; ++r) {
        float v = acc[t][r] + bb;
        if (proj == 0) v *= SCALE;
        base[(size_t)(p0 + r) * HD] = (__bf16)v;
      }
    }
  }
}

// ---------------------------------------------------------------------------
// Phase 2: flash attention.  grid=(64, NH, BATCH) block=128 (4 waves).
//  Each wave owns 16 queries; the block loops over keys in chunks of 32.
//  K (32x64) and V^T (64x32) chunks are DMA'd into LDS by the TDM, double
//  buffered so the next chunk streams in while all 4 waves compute.
//  Computes S^T = K·Q^T so the exp() tile is already in A-fragment layout
//  (queries in lanes, keys in VGPRs) for the P×V accumulation.
// ---------------------------------------------------------------------------
__global__ __launch_bounds__(128) void attn_kernel(
    const __bf16* __restrict__ Qb, const __bf16* __restrict__ Kb,
    const __bf16* __restrict__ Vt, __bf16* __restrict__ Ob) {
  const int tid  = threadIdx.x;
  const int lane = tid & 31;
  const int wave = tid >> 5;
  const int l16  = lane & 15;
  const int koff = (lane < 16) ? 0 : 8;
  const int rsel = (lane >> 4) << 3;

  const int b  = blockIdx.z;
  const int h  = blockIdx.y;
  const int q0 = blockIdx.x * 64 + wave * 16;

  const __bf16* Qh = Qb + ((size_t)(b * NH + h)) * SEQ * HD;
  const __bf16* Kh = Kb + ((size_t)(b * NH + h)) * SEQ * HD;
  const __bf16* Vh = Vt + ((size_t)(b * NH + h)) * HD * SEQ;

  // Double-buffered LDS staging: K chunk 32x64, V^T chunk 64x32 (4 KB each).
  __shared__ __attribute__((aligned(16))) __bf16 kbuf[2][32 * HD];
  __shared__ __attribute__((aligned(16))) __bf16 vbuf[2][HD * 32];
  unsigned int kOff[2], vOff[2];
  kOff[0] = (unsigned int)(unsigned long long)&kbuf[0][0];
  kOff[1] = (unsigned int)(unsigned long long)&kbuf[1][0];
  vOff[0] = (unsigned int)(unsigned long long)&vbuf[0][0];
  vOff[1] = (unsigned int)(unsigned long long)&vbuf[1][0];

  // Q^T B-fragments (lane = query, K-dim = d): loaded once from global.
  const __bf16* qrow = Qh + (size_t)(q0 + l16) * HD;
  v16bf bq0 = load_frag16(qrow, koff);        // d 0..31
  v16bf bq1 = load_frag16(qrow + 32, koff);   // d 32..63

  v8f o[4];
#pragma unroll
  for (int c = 0; c < 4; ++c)
#pragma unroll
    for (int r = 0; r < 8; ++r) o[c][r] = 0.0f;
  float mrun = -3.402823466e38f, lrun = 0.0f;

  // Prologue: kick off DMA of chunk 0.
  if (wave == 0) {
    tdm_load_2d_bf16(kOff[0], Kh, HD, SEQ, HD, 32, HD);       // K rows j0..j0+31
    tdm_load_2d_bf16(vOff[0], Vh, SEQ, HD, 32, HD, SEQ);      // V^T cols j0..+31
  }

  int cur = 0;
  for (int j0 = 0; j0 < SEQ; j0 += 32) {
    if (wave == 0) __builtin_amdgcn_s_wait_tensorcnt(0);
    __syncthreads();  // chunk `cur` visible to all; buffer cur^1 fully drained
    if (wave == 0 && (j0 + 32) < SEQ) {  // stream next chunk into other buffer
      tdm_load_2d_bf16(kOff[cur ^ 1], Kh + (size_t)(j0 + 32) * HD,
                       HD, SEQ, HD, 32, HD);
      tdm_load_2d_bf16(vOff[cur ^ 1], Vh + (j0 + 32),
                       SEQ, HD, 32, HD, SEQ);
    }
    const __bf16* kL = &kbuf[cur][0];
    const __bf16* vL = &vbuf[cur][0];

    // --- S^T tiles: rows = keys (VGPRs), cols = queries (lanes) ---
    const __bf16* kb0 = kL + (size_t)l16 * HD;         // keys 0..15 (local)
    const __bf16* kb1 = kL + (size_t)(16 + l16) * HD;  // keys 16..31
    v8f st0, st1;
#pragma unroll
    for (int r = 0; r < 8; ++r) { st0[r] = 0.0f; st1[r] = 0.0f; }
    st0 = WMMA_BF16(load_frag16(kb0, koff),      bq0, st0);
    st0 = WMMA_BF16(load_frag16(kb0 + 32, koff), bq1, st0);
    st1 = WMMA_BF16(load_frag16(kb1, koff),      bq0, st1);
    st1 = WMMA_BF16(load_frag16(kb1 + 32, koff), bq1, st1);

    // --- online softmax: each lane holds 16 of its query's 32 key scores ---
    float cm = st0[0];
#pragma unroll
    for (int r = 1; r < 8; ++r) cm = fmaxf(cm, st0[r]);
#pragma unroll
    for (int r = 0; r < 8; ++r) cm = fmaxf(cm, st1[r]);
    cm = fmaxf(cm, __shfl_xor(cm, 16, 32));   // combine lane halves
    const float mnew  = fmaxf(mrun, cm);
    const float alpha = __expf(mrun - mnew);

    float p0[8], p1[8], ps = 0.0f;
#pragma unroll
    for (int r = 0; r < 8; ++r) {
      p0[r] = __expf(st0[r] - mnew);
      p1[r] = __expf(st1[r] - mnew);
      ps += p0[r] + p1[r];
    }
    ps += __shfl_xor(ps, 16, 32);
    lrun = lrun * alpha + ps;
    mrun = mnew;

    // Register-local repack into A-fragment (queries already in lanes).
    v16bf pa;
#pragma unroll
    for (int r = 0; r < 8; ++r) {
      pa[r]     = (__bf16)p0[r];
      pa[8 + r] = (__bf16)p1[r];
    }

    // Rescale O rows (rows live in VGPRs -> broadcast alpha per row).
    float ar[8];
#pragma unroll
    for (int r = 0; r < 8; ++r) ar[r] = __shfl(alpha, r + rsel, 32);
#pragma unroll
    for (int c = 0; c < 4; ++c)
#pragma unroll
      for (int r = 0; r < 8; ++r) o[c][r] *= ar[r];

    // O += P × V   (V^T LDS layout -> contiguous ds_load_b128 B-frags).
#pragma unroll
    for (int c = 0; c < 4; ++c) {
      const __bf16* vb = vL + (size_t)(c * 16 + l16) * 32;
      o[c] = WMMA_BF16(pa, load_frag16(vb, koff), o[c]);
    }
    cur ^= 1;
  }

  // Normalize by row sums and store O (bf16 row-major [b,p,h,d]).
  const float linv = 1.0f / lrun;
  float lr[8];
#pragma unroll
  for (int r = 0; r < 8; ++r) lr[r] = __shfl(linv, r + rsel, 32);
#pragma unroll
  for (int c = 0; c < 4; ++c) {
    const int col = h * HD + c * 16 + l16;
#pragma unroll
    for (int r = 0; r < 8; ++r) {
      const int row = q0 + r + rsel;
      Ob[((size_t)(b * SEQ + row)) * CH + col] = (__bf16)(o[c][r] * lr[r]);
    }
  }
}

// ---------------------------------------------------------------------------
// Phase 3: output projection out = O @ Wo + bo (fp32 out).
//  grid=(256, 4, 1) block=128. Same GEMM skeleton as phase 1, bf16 A source.
// ---------------------------------------------------------------------------
__global__ __launch_bounds__(128) void oproj_kernel(
    const __bf16* __restrict__ Ob, const float* __restrict__ Wo,
    const float* __restrict__ bo, float* __restrict__ out) {
  const int tid  = threadIdx.x;
  const int lane = tid & 31;
  const int wave = tid >> 5;
  const int l16  = lane & 15;
  const int koff = (lane < 16) ? 0 : 8;
  const int m0   = blockIdx.x * 64 + wave * 16;
  const int n0   = blockIdx.y * 64;

  __shared__ __attribute__((aligned(16))) __bf16 wt[64][32];

  v8f acc[4];
#pragma unroll
  for (int t = 0; t < 4; ++t)
#pragma unroll
    for (int r = 0; r < 8; ++r) acc[t][r] = 0.0f;

  const __bf16* arow = Ob + (size_t)(m0 + l16) * CH;

  for (int k0 = 0; k0 < CH; k0 += 32) {
    __syncthreads();
#pragma unroll
    for (int i = 0; i < 16; ++i) {
      int idx = tid + i * 128;
      int n = idx & 63, k = idx >> 6;
      wt[n][k] = (__bf16)Wo[(size_t)(k0 + k) * CH + n0 + n];
    }
    __syncthreads();

    v16bf a = load_frag16(arow + k0, koff);
#pragma unroll
    for (int t = 0; t < 4; ++t) {
      v16bf bfrag = load_frag16(&wt[t * 16 + l16][0], koff);
      acc[t] = WMMA_BF16(a, bfrag, acc[t]);
    }
  }

  const int rowbase = m0 + ((lane >> 4) << 3);
#pragma unroll
  for (int t = 0; t < 4; ++t) {
    const int   col = n0 + t * 16 + l16;
    const float bb  = bo[col];
#pragma unroll
    for (int r = 0; r < 8; ++r)
      out[(size_t)(rowbase + r) * CH + col] = acc[t][r] + bb;
  }
}

// ---------------------------------------------------------------------------
extern "C" void kernel_launch(void* const* d_in, const int* in_sizes, int n_in,
                              void* d_out, int out_size, void* d_ws, size_t ws_size,
                              hipStream_t stream) {
  const float* x  = (const float*)d_in[0];
  const float* Wq = (const float*)d_in[1];
  const float* bq = (const float*)d_in[2];
  const float* Wk = (const float*)d_in[3];
  const float* bk = (const float*)d_in[4];
  const float* Wv = (const float*)d_in[5];
  const float* bv = (const float*)d_in[6];
  const float* Wo = (const float*)d_in[7];
  const float* bo = (const float*)d_in[8];
  float* out = (float*)d_out;

  const size_t SLAB = (size_t)BATCH * SEQ * CH;  // 4.19M elems (bf16 -> 8 MB)
  __bf16* Qb = (__bf16*)d_ws;
  __bf16* Kb = Qb + SLAB;
  __bf16* Vt = Kb + SLAB;
  __bf16* Ob = Vt + SLAB;   // total 32 MB of workspace

  dim3 blk(128);
  qkv_kernel<<<dim3(256, 4, 3), blk, 0, stream>>>(x, Wq, bq, Wk, bk, Wv, bv,
                                                  Qb, Kb, Vt);
  attn_kernel<<<dim3(SEQ / 64, NH, BATCH), blk, 0, stream>>>(Qb, Kb, Vt, Ob);
  oproj_kernel<<<dim3(256, 4, 1), blk, 0, stream>>>(Ob, Wo, bo, out);
}
